// ProofOptimizationGNN_16879221473285
// MI455X (gfx1250) — matrix operand order repaired and checked by reading, not compiled, not run on hardware
//
#include <hip/hip_runtime.h>
#include <hip/hip_bf16.h>

// ---------------- problem constants ----------------
#define NN 10000     // nodes
#define NE 320000    // edges
#define NF (NE + NN) // edges + self loops
#define NODE_F 64
#define EDGE_F 32
#define HD 256       // hidden
#define NL 6         // layers
#define HEADS 8
#define OUTC 32      // HD/HEADS
#define NG 16        // graphs

#define CDIV(a, b) (((a) + (b) - 1) / (b))

typedef __bf16 bf16;
typedef __attribute__((ext_vector_type(16))) __bf16 v16bf;
typedef __attribute__((ext_vector_type(8))) float v8f;

union FragBF {
    v16bf v;
    uint4 q[2];
};

// order-preserving float<->uint encoding for atomicMax (memset-0 == -inf)
__device__ __forceinline__ unsigned fenc(float f) {
    unsigned u = __float_as_uint(f);
    return (u & 0x80000000u) ? ~u : (u | 0x80000000u);
}
__device__ __forceinline__ float fdec(unsigned u) {
    return (u & 0x80000000u) ? __uint_as_float(u & 0x7fffffffu) : __uint_as_float(~u);
}
__device__ __forceinline__ float gelu_exact(float x) {
    return 0.5f * x * (1.0f + erff(x * 0.70710678118654752440f));
}

// ---------------- conversion ----------------
__global__ void k_cvt_bf16(const float* __restrict__ s, bf16* __restrict__ d, int n) {
    int t = blockIdx.x * blockDim.x + threadIdx.x;
    if (t < n) d[t] = (bf16)s[t];
}

// ---------------- attention-weight folding ----------------
// V[l][hd][c] = sum_o lin_edge_w[l][(hd*32+o)*HD + c] * att_edge[l][hd*32+o]
__global__ void k_fold_V(const float* __restrict__ lin_edge_w,
                         const float* __restrict__ att_edge,
                         float* __restrict__ V) {
    int t = blockIdx.x * blockDim.x + threadIdx.x;
    if (t >= NL * HEADS * HD) return;
    int c = t & (HD - 1);
    int hd = (t >> 8) & (HEADS - 1);
    int l = t >> 11;
    const float* W = lin_edge_w + (size_t)l * HD * HD + (size_t)(hd * OUTC) * HD + c;
    const float* a = att_edge + l * HD + hd * OUTC;
    float s = 0.f;
#pragma unroll
    for (int o = 0; o < OUTC; ++o) s += W[(size_t)o * HD] * a[o];
    V[t] = s;
}

// M[l][hd][f] = sum_c V[l][hd][c] * edge_emb_w[c*EDGE_F+f];  bias0[l][hd] = sum_c eeb[c]*V
__global__ void k_fold_M(const float* __restrict__ V,
                         const float* __restrict__ edge_emb_w,
                         const float* __restrict__ edge_emb_b,
                         float* __restrict__ M, float* __restrict__ bias0) {
    int t = blockIdx.x * blockDim.x + threadIdx.x;
    int nM = NL * HEADS * EDGE_F;
    if (t < nM) {
        int f = t & (EDGE_F - 1);
        int hd = (t >> 5) & (HEADS - 1);
        int l = t / (HEADS * EDGE_F);
        const float* v = V + (l * HEADS + hd) * HD;
        float s = 0.f;
        for (int c = 0; c < HD; ++c) s += v[c] * edge_emb_w[c * EDGE_F + f];
        M[t] = s;
    } else if (t < nM + NL * HEADS) {
        int j = t - nM;
        const float* v = V + j * HD;
        float s = 0.f;
        for (int c = 0; c < HD; ++c) s += v[c] * edge_emb_b[c];
        bias0[j] = s;
    }
}

// ---------------- graph preprocessing ----------------
__global__ void k_count(const int* __restrict__ dstv, float* __restrict__ cnt) {
    int e = blockIdx.x * blockDim.x + threadIdx.x;
    if (e < NE) atomicAdd(&cnt[dstv[e]], 1.0f);
}

// per-edge alpha-edge component for all 6 layers (48 values) + loop-attr segment sum
__global__ void k_alphaE(const float* __restrict__ edge_attr,
                         const float* __restrict__ M,
                         const float* __restrict__ bias0,
                         const int* __restrict__ dstv,
                         float* __restrict__ alphaE,
                         float* __restrict__ alphaL) {
    int e = blockIdx.x * blockDim.x + threadIdx.x;
    if (e >= NE) return;
    float ea[EDGE_F];
#pragma unroll
    for (int f = 0; f < EDGE_F; ++f) ea[f] = edge_attr[(size_t)e * EDGE_F + f];
    int d = dstv[e];
    for (int lh = 0; lh < NL * HEADS; ++lh) {
        const float* m = M + lh * EDGE_F;
        float s = bias0[lh];
#pragma unroll
        for (int f = 0; f < EDGE_F; ++f) s += ea[f] * m[f];
        alphaE[(size_t)e * 48 + lh] = s;
        atomicAdd(&alphaL[(size_t)d * 48 + lh], s);
    }
}

__global__ void k_alphaL_fin(float* __restrict__ alphaL, const float* __restrict__ cnt) {
    int t = blockIdx.x * blockDim.x + threadIdx.x;
    if (t >= NN * 48) return;
    alphaL[t] /= fmaxf(cnt[t / 48], 1.0f);
}

// ---------------- WMMA GEMM:  C[M,Ncol] = A[M,K](bf16) @ W[Ncol,K]^T(bf16) + bias ----------
// one wave per 16x64 output tile (4 N-tiles): A fragment loaded once per k-step and
// reused across 4 WMMAs -> 4 wmma : 10 b128-loads per iteration.
// fragments match ISA 7.12.2 16-bit layouts:
//   A 16x32: lane<16 row=l16 holds K{0..7,16..23}; lane>=16 holds K{8..15,24..31}
//   B 32x16: lane<16 col=l16 holds K{0..7,16..23}; lane>=16 holds K{8..15,24..31}
__global__ __launch_bounds__(32) void k_gemm_bf16(
    const bf16* __restrict__ A, const bf16* __restrict__ W,
    float* __restrict__ C, const float* __restrict__ bias, int K, int Ncol) {
    int lane = threadIdx.x & 31;
    int half = lane >> 4;
    int l16 = lane & 15;
    const bf16* arow = A + (size_t)(blockIdx.x * 16 + l16) * K;
    const bf16* wrow[4];
#pragma unroll
    for (int t = 0; t < 4; ++t)
        wrow[t] = W + (size_t)(blockIdx.y * 64 + t * 16 + l16) * K;
    v8f acc[4] = {{}, {}, {}, {}};
    for (int k0 = 0; k0 < K; k0 += 32) {
        FragBF a;
        a.q[0] = *(const uint4*)(arow + k0 + half * 8);
        a.q[1] = *(const uint4*)(arow + k0 + 16 + half * 8);
        FragBF b[4];
#pragma unroll
        for (int t = 0; t < 4; ++t) {
            b[t].q[0] = *(const uint4*)(wrow[t] + k0 + half * 8);
            b[t].q[1] = *(const uint4*)(wrow[t] + k0 + 16 + half * 8);
        }
#pragma unroll
        for (int t = 0; t < 4; ++t)
            acc[t] = __builtin_amdgcn_wmma_f32_16x16x32_bf16(false, a.v, false, b[t].v,
                                                             (short)0, acc[t], false, false);
    }
    int m0 = blockIdx.x * 16 + half * 8;
#pragma unroll
    for (int t = 0; t < 4; ++t) {
        int col = blockIdx.y * 64 + t * 16 + l16;
        float bb = bias ? bias[col] : 0.f;
#pragma unroll
        for (int r = 0; r < 8; ++r) C[(size_t)(m0 + r) * Ncol + col] = acc[t][r] + bb;
    }
}

// ---------------- per-node attention coefficients from xw ----------------
__global__ void k_attcoef(const float* __restrict__ xw,
                          const float* __restrict__ asrc_w,
                          const float* __restrict__ adst_w,
                          float* __restrict__ asrc, float* __restrict__ adst) {
    int t = blockIdx.x * blockDim.x + threadIdx.x;
    if (t >= NN * HEADS) return;
    int h = t & (HEADS - 1), n = t >> 3;
    const float* row = xw + (size_t)n * HD + h * OUTC;
    float s = 0.f, d = 0.f;
#pragma unroll
    for (int o = 0; o < OUTC; ++o) {
        s += row[o] * asrc_w[h * OUTC + o];
        d += row[o] * adst_w[h * OUTC + o];
    }
    asrc[t] = s;
    adst[t] = d;
}

// ---------------- edge pass 1: alpha = leaky(asrc+adst+comp); segment max ----------
__global__ void k_pass_max(const float* __restrict__ asrc, const float* __restrict__ adst,
                           const float* __restrict__ alphaE, const float* __restrict__ alphaL,
                           const int* __restrict__ srcv, const int* __restrict__ dstv,
                           unsigned* __restrict__ amax, float* __restrict__ ealpha, int l) {
    int i = blockIdx.x * blockDim.x + threadIdx.x;
    if (i >= NF) return;
    int s, d;
    const float* comp;
    if (i < NE) { s = srcv[i]; d = dstv[i]; comp = alphaE + (size_t)i * 48 + l * 8; }
    else        { s = d = i - NE;           comp = alphaL + (size_t)(i - NE) * 48 + l * 8; }
#pragma unroll
    for (int h = 0; h < HEADS; ++h) {
        float a = asrc[s * HEADS + h] + adst[d * HEADS + h] + comp[h];
        if (a < 0.f) a *= 0.2f;  // leaky_relu slope
        ealpha[(size_t)i * HEADS + h] = a;
        atomicMax(&amax[d * HEADS + h], fenc(a));
    }
}

// ---------------- edge pass 2: e = exp(alpha - amax); segment sum ----------------
__global__ void k_pass_exp(const float* __restrict__ ealpha, const unsigned* __restrict__ amax,
                           const int* __restrict__ dstv,
                           float* __restrict__ ee, float* __restrict__ denom) {
    int i = blockIdx.x * blockDim.x + threadIdx.x;
    if (i >= NF) return;
    int d = (i < NE) ? dstv[i] : i - NE;
#pragma unroll
    for (int h = 0; h < HEADS; ++h) {
        float m = fdec(amax[d * HEADS + h]);
        float e = expf(ealpha[(size_t)i * HEADS + h] - m);
        ee[(size_t)i * HEADS + h] = e;
        atomicAdd(&denom[d * HEADS + h], e);
    }
}

// ---------------- edge pass 3: weighted scatter of xw[src] (one wave / edge) ----------
__global__ void k_pass_agg(const float* __restrict__ ee, const float* __restrict__ denom,
                           const float* __restrict__ xw,
                           const int* __restrict__ srcv, const int* __restrict__ dstv,
                           float* __restrict__ agg) {
    int i = (blockIdx.x * blockDim.x + threadIdx.x) >> 5;
    if (i >= NF) return;
    int lane = threadIdx.x & 31;
    int s, d;
    if (i < NE) { s = srcv[i]; d = dstv[i]; } else { s = d = i - NE; }
    const float* xr = xw + (size_t)s * HD;
    __builtin_prefetch(xr + lane, 0, 1);  // global_prefetch_b8 on gathered row
#pragma unroll
    for (int h = 0; h < HEADS; ++h) {
        float w = ee[(size_t)i * HEADS + h] / (denom[d * HEADS + h] + 1e-16f);
        atomicAdd(&agg[(size_t)d * HD + h * OUTC + lane], xr[h * OUTC + lane] * w);
    }
}

// ---------------- node epilogue: +bias, LayerNorm, GELU, residual (wave32 reduce) -----
__global__ void k_node_ep(const float* __restrict__ agg, const float* __restrict__ cb,
                          const float* __restrict__ lg, const float* __restrict__ lb,
                          const float* __restrict__ hin, float* __restrict__ hout,
                          bf16* __restrict__ hb, int residual) {
    int n = (blockIdx.x * blockDim.x + threadIdx.x) >> 5;
    if (n >= NN) return;
    int lane = threadIdx.x & 31;
    float v[8];
    float s = 0.f;
#pragma unroll
    for (int j = 0; j < 8; ++j) {
        int c = j * 32 + lane;
        v[j] = agg[(size_t)n * HD + c] + cb[c];
        s += v[j];
    }
#pragma unroll
    for (int m = 16; m; m >>= 1) s += __shfl_xor(s, m, 32);
    float mu = s * (1.0f / HD);
    float var = 0.f;
#pragma unroll
    for (int j = 0; j < 8; ++j) { float dd = v[j] - mu; var += dd * dd; }
#pragma unroll
    for (int m = 16; m; m >>= 1) var += __shfl_xor(var, m, 32);
    float rs = rsqrtf(var * (1.0f / HD) + 1e-5f);
#pragma unroll
    for (int j = 0; j < 8; ++j) {
        int c = j * 32 + lane;
        float y = lg[c] * (v[j] - mu) * rs + lb[c];
        y = gelu_exact(y);
        if (residual) y += hin[(size_t)n * HD + c];
        hout[(size_t)n * HD + c] = y;
        hb[(size_t)n * HD + c] = (bf16)y;
    }
}

// ---------------- pooling ----------------
__global__ void k_pool(const float* __restrict__ h, const int* __restrict__ batch,
                       float* __restrict__ psum, float* __restrict__ pcnt) {
    int t = blockIdx.x * blockDim.x + threadIdx.x;
    if (t >= NN * HD) return;
    int n = t >> 8, c = t & 255;
    int g = batch[n];
    atomicAdd(&psum[g * HD + c], h[t]);
    if (c == 0) atomicAdd(&pcnt[g], 1.0f);
}

// ---------------- graph MLP heads (tiny: 5.3 MFLOP total, one block) ----------------
__global__ __launch_bounds__(512) void k_head(
    const float* __restrict__ psum, const float* __restrict__ pcnt,
    const float* pw1, const float* pb1, const float* pw2, const float* pb2,
    const float* pw3, const float* pb3,
    const float* mw1, const float* mb1, const float* mw2, const float* mb2,
    const float* mw3, const float* mb3, float* __restrict__ out) {
    __shared__ float sx[2 * HD], s1[HD], s2[HD / 2];
    int t = threadIdx.x;
    for (int g = 0; g < NG; ++g) {
        if (t < 2 * HD) {
            float c = fmaxf(pcnt[g], 1.0f);
            sx[t] = (t < HD) ? psum[g * HD + t] / c : psum[g * HD + (t - HD)];
        }
        __syncthreads();
        for (int head = 0; head < 2; ++head) {
            const float* w1 = head ? mw1 : pw1; const float* b1 = head ? mb1 : pb1;
            const float* w2 = head ? mw2 : pw2; const float* b2 = head ? mb2 : pb2;
            const float* w3 = head ? mw3 : pw3; const float* b3 = head ? mb3 : pb3;
            if (t < HD) {
                float a = b1[t];
                for (int k = 0; k < 2 * HD; ++k) a += sx[k] * w1[t * 2 * HD + k];
                s1[t] = gelu_exact(a);
            }
            __syncthreads();
            if (t < HD / 2) {
                float a = b2[t];
                for (int k = 0; k < HD; ++k) a += s1[k] * w2[t * HD + k];
                s2[t] = gelu_exact(a);
            }
            __syncthreads();
            if (t < 3) {
                float a = b3[t];
                for (int k = 0; k < HD / 2; ++k) a += s2[k] * w3[t * (HD / 2) + k];
                out[head * (NG * 3) + g * 3 + t] = 1.0f / (1.0f + expf(-a));
            }
            __syncthreads();
        }
    }
}

// =====================================================================
extern "C" void kernel_launch(void* const* d_in, const int* in_sizes, int n_in,
                              void* d_out, int out_size, void* d_ws, size_t ws_size,
                              hipStream_t stream) {
    (void)in_sizes; (void)n_in; (void)out_size; (void)ws_size;
    const float* x          = (const float*)d_in[0];
    const int*   edge_index = (const int*)d_in[1];
    const float* edge_attr  = (const float*)d_in[2];
    const int*   batch      = (const int*)d_in[3];
    const float* node_w     = (const float*)d_in[4];
    const float* node_b     = (const float*)d_in[5];
    const float* edge_emb_w = (const float*)d_in[6];
    const float* edge_emb_b = (const float*)d_in[7];
    const float* lin_w      = (const float*)d_in[8];
    const float* att_src    = (const float*)d_in[9];
    const float* att_dst    = (const float*)d_in[10];
    const float* att_edge   = (const float*)d_in[11];
    const float* lin_edge_w = (const float*)d_in[12];
    const float* conv_b     = (const float*)d_in[13];
    const float* ln_g       = (const float*)d_in[14];
    const float* ln_b       = (const float*)d_in[15];
    const float* pw1 = (const float*)d_in[16]; const float* pb1 = (const float*)d_in[17];
    const float* pw2 = (const float*)d_in[18]; const float* pb2 = (const float*)d_in[19];
    const float* pw3 = (const float*)d_in[20]; const float* pb3 = (const float*)d_in[21];
    const float* mw1 = (const float*)d_in[22]; const float* mb1 = (const float*)d_in[23];
    const float* mw2 = (const float*)d_in[24]; const float* mb2 = (const float*)d_in[25];
    const float* mw3 = (const float*)d_in[26]; const float* mb3 = (const float*)d_in[27];
    const int* srcv = edge_index;
    const int* dstv = edge_index + NE;
    float* out = (float*)d_out;

    // -------- workspace carve-out --------
    char* wp = (char*)d_ws;
    auto alloc = [&](size_t bytes) -> void* {
        void* p = (void*)wp;
        wp += (bytes + 255) & ~(size_t)255;
        return p;
    };
    float*    h0     = (float*)alloc((size_t)NN * HD * 4);
    float*    h1     = (float*)alloc((size_t)NN * HD * 4);
    bf16*     hb     = (bf16*)alloc((size_t)NN * HD * 2);
    bf16*     xb     = (bf16*)alloc((size_t)NN * NODE_F * 2);
    float*    xw     = (float*)alloc((size_t)NN * HD * 4);
    float*    asrc   = (float*)alloc((size_t)NN * HEADS * 4);
    float*    adst   = (float*)alloc((size_t)NN * HEADS * 4);
    float*    alphaE = (float*)alloc((size_t)NE * 48 * 4);
    float*    alphaL = (float*)alloc((size_t)NN * 48 * 4);
    float*    cnt    = (float*)alloc((size_t)NN * 4);
    unsigned* amax   = (unsigned*)alloc((size_t)NN * HEADS * 4);
    float*    denom  = (float*)alloc((size_t)NN * HEADS * 4);
    float*    ealpha = (float*)alloc((size_t)NF * HEADS * 4);
    float*    ee     = (float*)alloc((size_t)NF * HEADS * 4);
    float*    agg    = (float*)alloc((size_t)NN * HD * 4);
    float*    Vm     = (float*)alloc((size_t)NL * HEADS * HD * 4);
    float*    Mm     = (float*)alloc((size_t)NL * HEADS * EDGE_F * 4);
    float*    bias0  = (float*)alloc((size_t)NL * HEADS * 4);
    bf16*     wb_nd  = (bf16*)alloc((size_t)HD * NODE_F * 2);
    bf16*     wb_lin = (bf16*)alloc((size_t)NL * HD * HD * 2);
    float*    psum   = (float*)alloc((size_t)NG * HD * 4);
    float*    pcnt   = (float*)alloc((size_t)NG * 4);

    // -------- one-time preprocessing --------
    hipMemsetAsync(cnt, 0, (size_t)NN * 4, stream);
    hipMemsetAsync(alphaL, 0, (size_t)NN * 48 * 4, stream);

    k_cvt_bf16<<<CDIV(NN * NODE_F, 256), 256, 0, stream>>>(x, xb, NN * NODE_F);
    k_cvt_bf16<<<CDIV(HD * NODE_F, 256), 256, 0, stream>>>(node_w, wb_nd, HD * NODE_F);
    k_cvt_bf16<<<CDIV(NL * HD * HD, 256), 256, 0, stream>>>(lin_w, wb_lin, NL * HD * HD);

    k_fold_V<<<CDIV(NL * HEADS * HD, 256), 256, 0, stream>>>(lin_edge_w, att_edge, Vm);
    k_fold_M<<<CDIV(NL * HEADS * EDGE_F + NL * HEADS, 256), 256, 0, stream>>>(
        Vm, edge_emb_w, edge_emb_b, Mm, bias0);
    k_count<<<CDIV(NE, 256), 256, 0, stream>>>(dstv, cnt);
    k_alphaE<<<CDIV(NE, 256), 256, 0, stream>>>(edge_attr, Mm, bias0, dstv, alphaE, alphaL);
    k_alphaL_fin<<<CDIV(NN * 48, 256), 256, 0, stream>>>(alphaL, cnt);

    // node embedding: h0 = x @ node_w.T + node_b   (WMMA, K=64, 16x64 tiles)
    k_gemm_bf16<<<dim3(NN / 16, HD / 64), 32, 0, stream>>>(xb, wb_nd, h0, node_b, NODE_F, HD);
    k_cvt_bf16<<<CDIV(NN * HD, 256), 256, 0, stream>>>(h0, hb, NN * HD);

    // -------- GAT layers --------
    float* hin = h0;
    float* hout = h1;
    for (int l = 0; l < NL; ++l) {
        hipMemsetAsync(amax, 0, (size_t)NN * HEADS * 4, stream);   // 0 == encoded -inf
        hipMemsetAsync(denom, 0, (size_t)NN * HEADS * 4, stream);
        hipMemsetAsync(agg, 0, (size_t)NN * HD * 4, stream);

        // xw = h @ lin_w[l].T   (WMMA, K=256, 16x64 tiles with A-fragment reuse)
        k_gemm_bf16<<<dim3(NN / 16, HD / 64), 32, 0, stream>>>(
            hb, wb_lin + (size_t)l * HD * HD, xw, nullptr, HD, HD);

        k_attcoef<<<CDIV(NN * HEADS, 256), 256, 0, stream>>>(
            xw, att_src + l * HEADS * OUTC, att_dst + l * HEADS * OUTC, asrc, adst);

        k_pass_max<<<CDIV(NF, 256), 256, 0, stream>>>(
            asrc, adst, alphaE, alphaL, srcv, dstv, amax, ealpha, l);
        k_pass_exp<<<CDIV(NF, 256), 256, 0, stream>>>(ealpha, amax, dstv, ee, denom);
        k_pass_agg<<<CDIV(NF * 32, 256), 256, 0, stream>>>(ee, denom, xw, srcv, dstv, agg);

        k_node_ep<<<CDIV(NN * 32, 256), 256, 0, stream>>>(
            agg, conv_b + l * HD, ln_g + l * HD, ln_b + l * HD, hin, hout, hb, l > 0);

        float* tmp = hin; hin = hout; hout = tmp;  // hin now holds latest h
    }

    // -------- pooling + MLP heads --------
    hipMemsetAsync(psum, 0, (size_t)NG * HD * 4, stream);
    hipMemsetAsync(pcnt, 0, (size_t)NG * 4, stream);
    k_pool<<<CDIV(NN * HD, 256), 256, 0, stream>>>(hin, batch, psum, pcnt);
    k_head<<<1, 512, 0, stream>>>(psum, pcnt, pw1, pb1, pw2, pb2, pw3, pb3,
                                  mw1, mb1, mw2, mb2, mw3, mb3, out);
}
